// MoE_11081015623718
// MI455X (gfx1250) — matrix operand order repaired and checked by reading, not compiled, not run on hardware
//
#include <hip/hip_runtime.h>

#define T_TOK 2048
#define D_DIM 1024
#define F_DIM 4096
#define E_EXP 8

// ---------------------------------------------------------------------------
// native vector types (avoid HIP's constructor-laden vector structs in casts)
// ---------------------------------------------------------------------------
typedef __attribute__((ext_vector_type(16))) __bf16   v16bf;
typedef __attribute__((ext_vector_type(8)))  float    v8f;
typedef __attribute__((ext_vector_type(4)))  float    f32x4;
typedef __attribute__((ext_vector_type(4)))  unsigned u32x4;

struct B32x2 { u32x4 lo, hi; };

static __device__ __forceinline__ unsigned pk_bf16(float lo, float hi) {
  unsigned short l = __builtin_bit_cast(unsigned short, (__bf16)lo);
  unsigned short h = __builtin_bit_cast(unsigned short, (__bf16)hi);
  return ((unsigned)h << 16) | (unsigned)l;
}

static __device__ __forceinline__ v16bf make_frag(u32x4 lo, u32x4 hi) {
  B32x2 t{lo, hi};
  return __builtin_bit_cast(v16bf, t);
}

// CDNA5 async global->LDS copy (ASYNCcnt-tracked).  lds_off is the wave-
// relative LDS byte offset (low 32 bits of a generic pointer to __shared__).
static __device__ __forceinline__ void async_copy_b128(unsigned lds_off,
                                                       const void* gaddr) {
  asm volatile("global_load_async_to_lds_b128 %0, %1, off"
               :: "v"(lds_off), "v"(gaddr) : "memory");
}
static __device__ __forceinline__ void wait_async_le1() {
  asm volatile("s_wait_asynccnt 0x1" ::: "memory");
}

// ---------------------------------------------------------------------------
// Workspace layout (bytes):
//   [0)        counts[8]
//   [64)       base[9]             (32-aligned exclusive prefix of counts)
//   [256)      rows[E*T] int       (token id per expert slot)
//   [65792)    wts [E*T] float     (combine weight per expert slot)
//   [135168)   H   [4353 * F] bf16 (SwiGLU activations + 1 slack row)
// total ~35.8 MB
// ---------------------------------------------------------------------------
#define WS_OFF_COUNTS 0
#define WS_OFF_BASE   64
#define WS_OFF_ROWS   256
#define WS_OFF_WTS    (256 + 65536)
#define WS_OFF_H      135168

// ---------------------------------------------------------------------------
// Kernel 1: zero output accumulator + expert counters
// ---------------------------------------------------------------------------
__global__ void moe_init(float* __restrict__ out, int* __restrict__ counts) {
  size_t i = (size_t)blockIdx.x * 256 + threadIdx.x;
  f32x4 z = {0.f, 0.f, 0.f, 0.f};
  ((f32x4*)out)[i] = z;
  if (blockIdx.x == 0 && threadIdx.x < E_EXP) counts[threadIdx.x] = 0;
}

// ---------------------------------------------------------------------------
// Kernel 2: router — logits, top-2, softmax, bin tokens per expert
// ---------------------------------------------------------------------------
__global__ __launch_bounds__(128) void moe_router(
    const float* __restrict__ x, const float* __restrict__ rw,
    int* __restrict__ counts, int* __restrict__ rows, float* __restrict__ wts) {
  int t = blockIdx.x, tid = threadIdx.x;
  const float* xr = x + (size_t)t * D_DIM;
  float acc[E_EXP] = {};
  #pragma unroll
  for (int i = 0; i < D_DIM / 128; ++i) {
    int d = i * 128 + tid;
    float xv = xr[d];
    const f32x4* r4 = (const f32x4*)(rw + (size_t)d * E_EXP);
    f32x4 a = r4[0], b = r4[1];
    acc[0] += xv * a.x; acc[1] += xv * a.y; acc[2] += xv * a.z; acc[3] += xv * a.w;
    acc[4] += xv * b.x; acc[5] += xv * b.y; acc[6] += xv * b.z; acc[7] += xv * b.w;
  }
  #pragma unroll
  for (int off = 16; off >= 1; off >>= 1)
    #pragma unroll
    for (int e = 0; e < E_EXP; ++e) acc[e] += __shfl_down(acc[e], off, 32);

  __shared__ float part[4][E_EXP];
  int wave = tid >> 5, lane = tid & 31;
  if (lane == 0)
    #pragma unroll
    for (int e = 0; e < E_EXP; ++e) part[wave][e] = acc[e];
  __syncthreads();

  if (tid == 0) {
    float v[E_EXP];
    #pragma unroll
    for (int e = 0; e < E_EXP; ++e)
      v[e] = part[0][e] + part[1][e] + part[2][e] + part[3][e];
    int i0 = 0; float v0 = v[0];
    #pragma unroll
    for (int e = 1; e < E_EXP; ++e) if (v[e] > v0) { v0 = v[e]; i0 = e; }
    int i1 = -1; float v1 = -3.4e38f;
    #pragma unroll
    for (int e = 0; e < E_EXP; ++e)
      if (e != i0 && v[e] > v1) { v1 = v[e]; i1 = e; }
    float e1 = __expf(v1 - v0);
    float s  = 1.f + e1;
    float w0 = 1.f / s, w1 = e1 / s;
    int s0 = atomicAdd(&counts[i0], 1);
    rows[i0 * T_TOK + s0] = t; wts[i0 * T_TOK + s0] = w0;
    int s1 = atomicAdd(&counts[i1], 1);
    rows[i1 * T_TOK + s1] = t; wts[i1 * T_TOK + s1] = w1;
  }
}

// ---------------------------------------------------------------------------
// Kernel 3: tiny scan for 32-row-aligned per-expert bases into H
// ---------------------------------------------------------------------------
__global__ void moe_scan(const int* __restrict__ counts, int* __restrict__ base) {
  if (blockIdx.x == 0 && threadIdx.x == 0) {
    int b = 0;
    for (int e = 0; e < E_EXP; ++e) { base[e] = b; b += (counts[e] + 31) & ~31; }
    base[E_EXP] = b;
  }
}

// ---------------------------------------------------------------------------
// LDS staging: 32(K) x 256(N) fp32 tile (row-major, leading dim ld) from
// global -> transposed bf16 [256 N rows][K pairs], row stride 40 elems (80 B,
// 16B-aligned so B fragments are two ds_load_b128 per lane).  Prefetches the
// next k-step's tile while this one streams in.
// ---------------------------------------------------------------------------
static __device__ __forceinline__ void stage_tile(
    const float* __restrict__ src, int ld, __bf16* dst, int tid) {
  unsigned* d = (unsigned*)dst;
  #pragma unroll
  for (int rep = 0; rep < 8; ++rep) {
    int slot = rep * 128 + tid;
    int kp = slot & 15;          // k pair: k = 2*kp
    int c  = (slot >> 4) * 4;    // column base
    const float* p = src + (size_t)(2 * kp) * ld + c;
    __builtin_prefetch(p + (size_t)32 * ld, 0, 3);   // next K-tile, same cols
    f32x4 v0 = *(const f32x4*)p;
    f32x4 v1 = *(const f32x4*)(p + ld);
    d[(c + 0) * 20 + kp] = pk_bf16(v0.x, v1.x);
    d[(c + 1) * 20 + kp] = pk_bf16(v0.y, v1.y);
    d[(c + 2) * 20 + kp] = pk_bf16(v0.z, v1.z);
    d[(c + 3) * 20 + kp] = pk_bf16(v0.w, v1.w);
  }
}

// B fragment (K=32, 16 cols) from transposed LDS tile: lane(n,h) VGPR v holds
// K = 16h+2v, 16h+2v+1  -> bytes [32h,32h+16) and [32h+16,32h+32) of row n.
static __device__ __forceinline__ v16bf load_frag_b(
    const __bf16* tile, int row, int h) {
  const char* p = (const char*)tile + (size_t)row * 80 + h * 32;
  return make_frag(*(const u32x4*)p, *(const u32x4*)(p + 16));
}

// A fragment (16 rows, K=32) from a row-major bf16 tile (80 B row stride):
// lane(m,h) bytes [32h,32h+16) and [32h+16,32h+32) of row m.
static __device__ __forceinline__ v16bf load_frag_a(
    const char* rowptr, int h) {
  const char* p = rowptr + h * 32;
  return make_frag(*(const u32x4*)p, *(const u32x4*)(p + 16));
}

// ---------------------------------------------------------------------------
// Kernel 4: grouped gate+up GEMM with fused SwiGLU  -> H (bf16)
// grid (F/256, T/32, E), 128 threads = 4 waves; wave owns 64 F columns,
// block owns 32 token rows (two A tiles -> every B fragment feeds 2 WMMAs).
// K = D = 1024 -> 32 k-steps; 16 bf16 WMMAs per wave per k-step, 2 barriers.
// ---------------------------------------------------------------------------
__global__ __launch_bounds__(128) void moe_gate_up(
    const float* __restrict__ x, const float* __restrict__ wg,
    const float* __restrict__ wu, const int* __restrict__ counts,
    const int* __restrict__ base, const int* __restrict__ rows,
    __bf16* __restrict__ H) {
  int e = blockIdx.z;
  int cnt = counts[e];
  int m0 = blockIdx.y * 32;
  if (m0 >= cnt) return;
  int fc0 = blockIdx.x * 256;
  int a0 = base[e] + m0;

  __shared__ __bf16 Wgl[256 * 40];     // 20.5 KB staged gate tile
  __shared__ __bf16 Wul[256 * 40];     // 20.5 KB staged up tile
  __shared__ __bf16 Xl[32 * 40];       // 2.5 KB per-k-step x tile (32 rows)
  __shared__ int tokl[32];

  int tid = threadIdx.x;
  if (tid < 32) {
    int idx = m0 + tid;
    tokl[tid] = (idx < cnt) ? rows[e * T_TOK + idx] : rows[e * T_TOK + m0];
  }
  __syncthreads();

  // per-thread x staging slot: row = tid/4, 8 K-elements at (tid%4)*8
  int xrow = tid >> 2, xk8 = (tid & 3) * 8;
  const float* xsrc = x + (size_t)tokl[xrow] * D_DIM + xk8;
  u32x4* xdst = (u32x4*)((char*)Xl + (size_t)xrow * 80 + (tid & 3) * 16);

  int wave = tid >> 5, lane = tid & 31;
  int n = lane & 15, h = lane >> 4;

  v8f accg[2][4] = {};
  v8f accu[2][4] = {};

  for (int ks = 0; ks < D_DIM / 32; ++ks) {
    int K0 = ks * 32;
    __syncthreads();
    stage_tile(wg + ((size_t)e * D_DIM + K0) * F_DIM + fc0, F_DIM, Wgl, tid);
    stage_tile(wu + ((size_t)e * D_DIM + K0) * F_DIM + fc0, F_DIM, Wul, tid);
    {
      f32x4 v0 = *(const f32x4*)(xsrc + K0);
      f32x4 v1 = *(const f32x4*)(xsrc + K0 + 4);
      u32x4 pk = {pk_bf16(v0.x, v0.y), pk_bf16(v0.z, v0.w),
                  pk_bf16(v1.x, v1.y), pk_bf16(v1.z, v1.w)};
      *xdst = pk;
    }
    __syncthreads();

    v16bf A0 = load_frag_a((const char*)Xl + (size_t)n * 80, h);
    v16bf A1 = load_frag_a((const char*)Xl + (size_t)(n + 16) * 80, h);

    v16bf Bg[4];
    #pragma unroll
    for (int nt = 0; nt < 4; ++nt)
      Bg[nt] = load_frag_b(Wgl, wave * 64 + nt * 16 + n, h);
    #pragma unroll
    for (int nt = 0; nt < 4; ++nt) {
      accg[0][nt] = __builtin_amdgcn_wmma_f32_16x16x32_bf16(
          false, A0, false, Bg[nt], (short)0, accg[0][nt], false, false);
      accg[1][nt] = __builtin_amdgcn_wmma_f32_16x16x32_bf16(
          false, A1, false, Bg[nt], (short)0, accg[1][nt], false, false);
    }

    v16bf Bu[4];
    #pragma unroll
    for (int nt = 0; nt < 4; ++nt)
      Bu[nt] = load_frag_b(Wul, wave * 64 + nt * 16 + n, h);
    #pragma unroll
    for (int nt = 0; nt < 4; ++nt) {
      accu[0][nt] = __builtin_amdgcn_wmma_f32_16x16x32_bf16(
          false, A0, false, Bu[nt], (short)0, accu[0][nt], false, false);
      accu[1][nt] = __builtin_amdgcn_wmma_f32_16x16x32_bf16(
          false, A1, false, Bu[nt], (short)0, accu[1][nt], false, false);
    }
  }

  // SwiGLU + store H (bf16). C layout: VGPR r, lane(n,h) -> (M=r+8h, N=n).
  #pragma unroll
  for (int mt = 0; mt < 2; ++mt) {
    #pragma unroll
    for (int nt = 0; nt < 4; ++nt) {
      int f = fc0 + wave * 64 + nt * 16 + n;
      #pragma unroll
      for (int r = 0; r < 8; ++r) {
        int m = mt * 16 + r + 8 * h;
        float g = accg[mt][nt][r], u = accu[mt][nt][r];
        float sv = g * __builtin_amdgcn_rcpf(1.f + __expf(-g));
        H[(size_t)(a0 + m) * F_DIM + f] = (__bf16)(sv * u);
      }
    }
  }
}

// ---------------------------------------------------------------------------
// Kernel 5: grouped down GEMM + weighted scatter into out
// grid (D/256, T/32, E), 128 threads = 4 waves; wave owns 64 D columns,
// block owns 32 token rows.  A tiles (32x32 of H) are fetched one k-step
// ahead with double-buffered global_load_async_to_lds_b128 (ASYNCcnt).
// K = F = 4096 -> 128 k-steps; 8 bf16 WMMAs per wave per k-step.
// ---------------------------------------------------------------------------
__global__ __launch_bounds__(128) void moe_down(
    const float* __restrict__ wd, const int* __restrict__ counts,
    const int* __restrict__ base, const int* __restrict__ rows,
    const float* __restrict__ wts, const __bf16* __restrict__ H,
    float* __restrict__ out) {
  int e = blockIdx.z;
  int cnt = counts[e];
  int m0 = blockIdx.y * 32;
  if (m0 >= cnt) return;
  int dc0 = blockIdx.x * 256;
  int a0 = base[e] + m0;

  __shared__ __bf16 Wl[256 * 40];      // 20.5 KB staged W_down tile
  __shared__ __bf16 Hl[2 * 32 * 40];   // 5 KB double-buffered A tiles
  __shared__ int   tokl[32];
  __shared__ float wtl[32];

  int tid = threadIdx.x;
  if (tid < 32) {
    int idx = m0 + tid;
    bool vld = idx < cnt;
    tokl[tid] = vld ? rows[e * T_TOK + idx] : 0;
    wtl[tid]  = vld ? wts[e * T_TOK + idx] : 0.f;
  }

  // async A-tile copy: thread -> (row = tid/4, 16B segment tid%4); 32x64B tile
  int arow = tid >> 2, aseg = tid & 3;
  const char* hsrc = (const char*)H +
      ((size_t)(a0 + arow) * F_DIM) * 2 + (size_t)aseg * 16;
  unsigned hl0 = (unsigned)(size_t)(&Hl[0]) + arow * 80 + aseg * 16;

  async_copy_b128(hl0, hsrc);                    // prologue: tile ks=0 -> buf0

  int wave = tid >> 5, lane = tid & 31;
  int n = lane & 15, h = lane >> 4;

  v8f acc[2][4] = {};

  for (int ks = 0; ks < F_DIM / 32; ++ks) {
    int K0 = ks * 32;
    __syncthreads();
    stage_tile(wd + ((size_t)e * F_DIM + K0) * D_DIM + dc0, D_DIM, Wl, tid);
    // prefetch next A tile into the other buffer (last one lands in slack row)
    async_copy_b128(hl0 + ((ks + 1) & 1) * 2560, hsrc + (size_t)(ks + 1) * 64);
    wait_async_le1();                            // current A tile complete
    __syncthreads();

    const char* hb = (const char*)Hl + (ks & 1) * 2560;
    v16bf A0 = load_frag_a(hb + (size_t)n * 80, h);
    v16bf A1 = load_frag_a(hb + (size_t)(n + 16) * 80, h);

    v16bf B[4];
    #pragma unroll
    for (int nt = 0; nt < 4; ++nt)
      B[nt] = load_frag_b(Wl, wave * 64 + nt * 16 + n, h);
    #pragma unroll
    for (int nt = 0; nt < 4; ++nt) {
      acc[0][nt] = __builtin_amdgcn_wmma_f32_16x16x32_bf16(
          false, A0, false, B[nt], (short)0, acc[0][nt], false, false);
      acc[1][nt] = __builtin_amdgcn_wmma_f32_16x16x32_bf16(
          false, A1, false, B[nt], (short)0, acc[1][nt], false, false);
    }
  }

  // weighted combine: exactly two (commutative) atomic adds per out element
  #pragma unroll
  for (int mt = 0; mt < 2; ++mt) {
    #pragma unroll
    for (int nt = 0; nt < 4; ++nt) {
      int d = dc0 + wave * 64 + nt * 16 + n;
      #pragma unroll
      for (int r = 0; r < 8; ++r) {
        int m = mt * 16 + r + 8 * h;
        atomicAdd(&out[(size_t)tokl[m] * D_DIM + d], wtl[m] * acc[mt][nt][r]);
      }
    }
  }
}

// ---------------------------------------------------------------------------
extern "C" void kernel_launch(void* const* d_in, const int* in_sizes, int n_in,
                              void* d_out, int out_size, void* d_ws, size_t ws_size,
                              hipStream_t stream) {
  (void)in_sizes; (void)n_in; (void)out_size; (void)ws_size;
  const float* x  = (const float*)d_in[0];
  const float* rw = (const float*)d_in[1];
  const float* wg = (const float*)d_in[2];
  const float* wu = (const float*)d_in[3];
  const float* wd = (const float*)d_in[4];
  float* out = (float*)d_out;
  char* ws = (char*)d_ws;

  int*    counts = (int*)(ws + WS_OFF_COUNTS);
  int*    base   = (int*)(ws + WS_OFF_BASE);
  int*    rows   = (int*)(ws + WS_OFF_ROWS);
  float*  wts    = (float*)(ws + WS_OFF_WTS);
  __bf16* H      = (__bf16*)(ws + WS_OFF_H);

  moe_init  <<<dim3(T_TOK * D_DIM / 4 / 256), dim3(256), 0, stream>>>(out, counts);
  moe_router<<<dim3(T_TOK),                   dim3(128), 0, stream>>>(x, rw, counts, rows, wts);
  moe_scan  <<<dim3(1),                       dim3(32),  0, stream>>>(counts, base);
  moe_gate_up<<<dim3(F_DIM / 256, T_TOK / 32, E_EXP), dim3(128), 0, stream>>>(
      x, wg, wu, counts, base, rows, H);
  moe_down  <<<dim3(D_DIM / 256, T_TOK / 32, E_EXP), dim3(128), 0, stream>>>(
      wd, counts, base, rows, wts, H, out);
}
